// MultiHeadAttention_22050362097908
// MI455X (gfx1250) — matrix operand order repaired and checked by reading, not compiled, not run on hardware
//
#include <hip/hip_runtime.h>
#include <hip/hip_bf16.h>

// MHA forward for MI455X (gfx1250).
// All GEMMs on v_wmma_f32_16x16x32_bf16 (f32 accum).
// - Projections: double-buffered LDS ping-pong, register-staged global fetch
//   overlapped with WMMA compute, one barrier per K-step.
// - Attention: flash-style; K/V tiles double-buffered via Tensor Data Mover
//   (tensor_load_to_lds + s_wait_tensorcnt) when available; Q fragments
//   hoisted to registers; mask tile staged to LDS with b128 loads.
// B=2, S=2048, D_MODEL=1024, H=16, HEAD_DIM=64.

typedef __bf16 bf16;
typedef __attribute__((ext_vector_type(16))) __bf16 v16bf;
typedef __attribute__((ext_vector_type(8)))  float  v8f;
typedef __attribute__((ext_vector_type(4)))  unsigned int u32x4;
typedef __attribute__((ext_vector_type(8)))  int i32x8;
typedef __attribute__((ext_vector_type(4)))  int i32x4;

#define D_MODEL  1024
#define N_HEADS  16
#define HEAD_DIM 64
#define BATCH    2
#define SEQ      2048

#if defined(__gfx1250__) && __has_builtin(__builtin_amdgcn_tensor_load_to_lds) && \
    __has_builtin(__builtin_amdgcn_s_wait_tensorcnt)
#define HAVE_TDM 1
#else
#define HAVE_TDM 0
#endif

static __device__ __forceinline__ bf16 f2bf(float x) {
  unsigned u = __builtin_bit_cast(unsigned, x);
  unsigned r = (u + 0x7FFFu + ((u >> 16) & 1u)) >> 16;   // round-nearest-even
  unsigned short h = (unsigned short)r;
  return __builtin_bit_cast(bf16, h);
}

// Register staging for one 16-element row chunk (128-bit global loads).
template <typename T> struct StageReg;
template <> struct StageReg<float> {
  float4 v[4];
  __device__ __forceinline__ void fetch(const float* src) {
    const float4* s4 = (const float4*)src;
#pragma unroll
    for (int j = 0; j < 4; ++j) v[j] = s4[j];
  }
  __device__ __forceinline__ void store(bf16* dst) const {
#pragma unroll
    for (int j = 0; j < 4; ++j) {
      dst[j * 4 + 0] = f2bf(v[j].x); dst[j * 4 + 1] = f2bf(v[j].y);
      dst[j * 4 + 2] = f2bf(v[j].z); dst[j * 4 + 3] = f2bf(v[j].w);
    }
  }
};
template <> struct StageReg<bf16> {
  uint4 v[2];
  __device__ __forceinline__ void fetch(const bf16* src) {
    const uint4* s4 = (const uint4*)src;
    v[0] = s4[0]; v[1] = s4[1];
  }
  __device__ __forceinline__ void store(bf16* dst) const {
    uint4* d4 = (uint4*)dst;
    d4[0] = v[0]; d4[1] = v[1];
  }
};

static __device__ __forceinline__ void stage16(bf16* dst, const bf16* src) {
  const uint4* s4 = (const uint4*)src;
  uint4* d4 = (uint4*)dst;
  d4[0] = s4[0]; d4[1] = s4[1];
}

// reductions confined to the 16-lane halves of a wave32 (masks 1,2,4,8)
static __device__ __forceinline__ float half16_max(float x) {
#pragma unroll
  for (int m = 1; m < 16; m <<= 1) x = fmaxf(x, __shfl_xor(x, m, 32));
  return x;
}
static __device__ __forceinline__ float half16_sum(float x) {
#pragma unroll
  for (int m = 1; m < 16; m <<= 1) x += __shfl_xor(x, m, 32);
  return x;
}

#if HAVE_TDM
// One TDM 2-D tile load: gptr -> LDS[lds_addr], tile (d0 x d1) bf16,
// row stride stride0 elements. Descriptor per CDNA5 ISA §8.3/8.4:
// group0: count=1, lds_addr[63:32], global_addr[120:64], type=2[127:126]
// group1: data_size=1 (2B)[17:16], tensor_dim0[79:48], tensor_dim1[111:80],
//         tile_dim0[127:112], tile_dim1[143:128], dim0_stride[207:160]
static __device__ __forceinline__ void tdm_load_2d_bf16(
    unsigned lds_addr, const bf16* gptr, unsigned tile_d0, unsigned tile_d1,
    unsigned tens_d0, unsigned tens_d1, unsigned stride0)
{
  unsigned long long ga = (unsigned long long)(uintptr_t)gptr;
  u32x4 g0;
  g0[0] = 1u;                                                // count=1 (user)
  g0[1] = lds_addr;
  g0[2] = (unsigned)(ga & 0xFFFFFFFFu);
  g0[3] = (unsigned)((ga >> 32) & 0x01FFFFFFu) | (2u << 30); // addr hi | type=2
  i32x8 g1;
  g1[0] = 1 << 16;                                           // data_size=2B
  g1[1] = (int)((tens_d0 & 0xFFFFu) << 16);
  g1[2] = (int)(((tens_d0 >> 16) & 0xFFFFu) | ((tens_d1 & 0xFFFFu) << 16));
  g1[3] = (int)(((tens_d1 >> 16) & 0xFFFFu) | ((tile_d0 & 0xFFFFu) << 16));
  g1[4] = (int)(tile_d1 & 0xFFFFu);
  g1[5] = (int)stride0;                                      // dim0_stride lo32
  g1[6] = 0;
  g1[7] = 0;
  i32x4 z4 = {0, 0, 0, 0};
#if __has_include(<hip/amd_detail/amd_gfx1250_TDM.h>)
  i32x8 z8 = {0, 0, 0, 0, 0, 0, 0, 0};
  __builtin_amdgcn_tensor_load_to_lds(g0, g1, z4, z4, z8, 0);   // clang-23 form
#else
  __builtin_amdgcn_tensor_load_to_lds(g0, g1, z4, z4, 0);       // ROCm 7.2 form
#endif
}
#endif

// ---------------------------------------------------------------------------
// GEMM + bias: out = X[M,K] @ W[K,N] + bias[N]
// MODE 0: bf16 out to [b, h, s, d] (QKV projections); MODE 1: f32 out [M, N].
// 256 threads = 8 waves as 2x4; each wave: 64x32 region = 4x2 WMMA tiles.
// Software pipeline: fetch tile t+1 into registers while computing tile t
// from LDS buffer `cur`; store into buffer `1-cur`; one barrier per K-step.
// ---------------------------------------------------------------------------
template <typename TIN, int MODE>
__global__ __launch_bounds__(256) void gemm_bias_wmma(
    const TIN* __restrict__ X, const float* __restrict__ W,
    const float* __restrict__ bias, bf16* __restrict__ outb,
    float* __restrict__ outf, int M, int N, int K)
{
  constexpr int BM = 128, BN = 128, BK = 32;
  __shared__ bf16 As[2][BM][BK];        // 16 KB
  __shared__ bf16 Bs[2][BK][BN + 8];    // 17 KB

  const int tid  = threadIdx.x;
  const int lane = tid & 31;
  const int wave = tid >> 5;
  const int wm   = wave >> 2;
  const int wn   = wave & 3;
  const int m0   = blockIdx.y * BM;
  const int n0   = blockIdx.x * BN;
  const int l15  = lane & 15;
  const int half8 = (lane & 16) ? 8 : 0;
  const int kba  = (lane < 16) ? 0 : 8;   // unused; A frags also use kbb
  const int kbb  = (lane < 16) ? 0 : 16;  // B-frag K base

  const v8f vzero = {0.f, 0.f, 0.f, 0.f, 0.f, 0.f, 0.f, 0.f};
  v8f acc[4][2];
#pragma unroll
  for (int mt = 0; mt < 4; ++mt)
#pragma unroll
    for (int nt = 0; nt < 2; ++nt) acc[mt][nt] = vzero;

  const int arow  = tid >> 1;
  const int akoff = (tid & 1) * 16;
  const int brow  = tid >> 3;
  const int bcoff = (tid & 7) * 16;
  const int nIter = K / BK;

  StageReg<TIN>   ra;
  StageReg<float> rb;

  // prologue: tile 0 -> buffer 0
  ra.fetch(X + (size_t)(m0 + arow) * K + akoff);
  rb.fetch(W + (size_t)brow * N + n0 + bcoff);
  ra.store(&As[0][arow][akoff]);
  rb.store(&Bs[0][brow][bcoff]);
  __syncthreads();

  for (int t = 0; t < nIter; ++t) {
    const int cur = t & 1;
    if (t + 1 < nIter) {   // overlap next tile's global fetch with WMMAs
      const int k0n = (t + 1) * BK;
      ra.fetch(X + (size_t)(m0 + arow) * K + k0n + akoff);
      rb.fetch(W + (size_t)(k0n + brow) * N + n0 + bcoff);
    }

    v16bf a[4], b[2];
#pragma unroll
    for (int mt = 0; mt < 4; ++mt) {
      const int r = wm * 64 + mt * 16 + l15;
#pragma unroll
      for (int e = 0; e < 8; ++e) {
        a[mt][e]     = As[cur][r][kbb + e];
        a[mt][8 + e] = As[cur][r][kbb + 8 + e];
      }
    }
#pragma unroll
    for (int nt = 0; nt < 2; ++nt) {
      const int c = wn * 32 + nt * 16 + l15;
#pragma unroll
      for (int e = 0; e < 16; ++e) b[nt][e] = Bs[cur][kbb + e][c];
    }
#pragma unroll
    for (int mt = 0; mt < 4; ++mt)
#pragma unroll
      for (int nt = 0; nt < 2; ++nt)
        acc[mt][nt] = __builtin_amdgcn_wmma_f32_16x16x32_bf16(
            false, a[mt], false, b[nt], (short)0, acc[mt][nt], false, false);

    if (t + 1 < nIter) {
      ra.store(&As[1 - cur][arow][akoff]);
      rb.store(&Bs[1 - cur][brow][bcoff]);
      __syncthreads();   // publish next buffer (prev reads of it fenced at t-1)
    }
  }

#pragma unroll
  for (int mt = 0; mt < 4; ++mt) {
#pragma unroll
    for (int nt = 0; nt < 2; ++nt) {
      const int col = n0 + wn * 32 + nt * 16 + l15;
      const float bia = bias[col];
#pragma unroll
      for (int r = 0; r < 8; ++r) {
        const int row = m0 + wm * 64 + mt * 16 + r + half8;
        const float v = acc[mt][nt][r] + bia;
        if (MODE == 0) {
          const int b_ = row >> 11, s_ = row & (SEQ - 1);
          const int h_ = col >> 6,  d_ = col & (HEAD_DIM - 1);
          outb[((size_t)(b_ * N_HEADS + h_) * SEQ + s_) * HEAD_DIM + d_] = f2bf(v);
        } else {
          outf[(size_t)row * N + col] = v;
        }
      }
    }
  }
}

// ---------------------------------------------------------------------------
// Flash attention: one block = 128 query rows of one (b, h); 8 waves, each
// owning 16 query rows. K/V tiles double-buffered in LDS: TDM streams block
// i+1 while waves compute block i (s_wait_tensorcnt + barrier to publish).
// Q fragments hoisted to registers. Mask tile staged to LDS with b128 loads.
// LDS: Q 16K + 2xK 32K + 2xV 32K + P 35K + mask 18K ~= 133 KB (WGP: 320 KB).
// ---------------------------------------------------------------------------
__global__ __launch_bounds__(256) void attention_wmma(
    const bf16* __restrict__ qb, const bf16* __restrict__ kbuf,
    const bf16* __restrict__ vbuf, const unsigned char* __restrict__ mask,
    bf16* __restrict__ ctxw)
{
  __shared__ bf16 Qs[128][HEAD_DIM];
  __shared__ bf16 Ks[2][128][HEAD_DIM];
  __shared__ bf16 Vs[2][128][HEAD_DIM];
  __shared__ bf16 Ps[8][16][128 + 8];
  __shared__ unsigned char Ms[128][144];

  const int tid  = threadIdx.x;
  const int lane = tid & 31;
  const int wave = tid >> 5;
  const int l15  = lane & 15;
  const int half8 = (lane & 16) ? 8 : 0;
  const int kba  = (lane < 16) ? 0 : 8;
  const int kbb  = (lane < 16) ? 0 : 16;

  const int q0 = blockIdx.x * 128;
  const int bh = blockIdx.y;
  const int b  = bh >> 4;
  const int h  = bh & 15;
  const size_t head_base = (size_t)bh * SEQ * HEAD_DIM;
  const bf16* kbase_p = kbuf + head_base;
  const bf16* vbase_p = vbuf + head_base;
  constexpr int NKB = SEQ / 128;

  // ---- prologue: Q tile + K/V block 0 into buffer 0 ----
#if HAVE_TDM
  if (wave == 0) {
    tdm_load_2d_bf16((unsigned)(uintptr_t)&Qs[0][0],
                     qb + head_base + (size_t)q0 * HEAD_DIM,
                     HEAD_DIM, 128, HEAD_DIM, SEQ, HEAD_DIM);
    tdm_load_2d_bf16((unsigned)(uintptr_t)&Ks[0][0][0], kbase_p,
                     HEAD_DIM, 128, HEAD_DIM, SEQ, HEAD_DIM);
    tdm_load_2d_bf16((unsigned)(uintptr_t)&Vs[0][0][0], vbase_p,
                     HEAD_DIM, 128, HEAD_DIM, SEQ, HEAD_DIM);
    __builtin_amdgcn_s_wait_tensorcnt(0);
  }
#else
  {
    const int row  = tid >> 1;
    const int doff = (tid & 1) * 32;
    stage16(&Qs[row][doff],      qb + head_base + (size_t)(q0 + row) * HEAD_DIM + doff);
    stage16(&Qs[row][doff + 16], qb + head_base + (size_t)(q0 + row) * HEAD_DIM + doff + 16);
    stage16(&Ks[0][row][doff],      kbase_p + (size_t)row * HEAD_DIM + doff);
    stage16(&Ks[0][row][doff + 16], kbase_p + (size_t)row * HEAD_DIM + doff + 16);
    stage16(&Vs[0][row][doff],      vbase_p + (size_t)row * HEAD_DIM + doff);
    stage16(&Vs[0][row][doff + 16], vbase_p + (size_t)row * HEAD_DIM + doff + 16);
  }
#endif
  __syncthreads();

  // Q fragments are loop-invariant: hoist into registers once.
  v16bf aq[2];
  {
    const int qr = wave * 16 + l15;
#pragma unroll
    for (int d0i = 0; d0i < 2; ++d0i) {
      const int d0 = d0i * 32;
#pragma unroll
      for (int e = 0; e < 8; ++e) {
        aq[d0i][e]     = Qs[qr][d0 + kbb + e];
        aq[d0i][8 + e] = Qs[qr][d0 + kbb + 8 + e];
      }
    }
  }

  float m_run[8], l_run[8];
  const v8f vzero = {0.f, 0.f, 0.f, 0.f, 0.f, 0.f, 0.f, 0.f};
  v8f ctx[4];
#pragma unroll
  for (int r = 0; r < 8; ++r) { m_run[r] = -1e30f; l_run[r] = 0.f; }
#pragma unroll
  for (int nt = 0; nt < 4; ++nt) ctx[nt] = vzero;

  for (int i = 0; i < NKB; ++i) {
    const int cur = i & 1;
    const int kb0 = i * 128;

#if HAVE_TDM
    // Kick off TDM for the NEXT K/V block into the other buffer.
    if (i + 1 < NKB && wave == 0) {
      tdm_load_2d_bf16((unsigned)(uintptr_t)&Ks[1 - cur][0][0],
                       kbase_p + (size_t)(kb0 + 128) * HEAD_DIM,
                       HEAD_DIM, 128, HEAD_DIM, SEQ, HEAD_DIM);
      tdm_load_2d_bf16((unsigned)(uintptr_t)&Vs[1 - cur][0][0],
                       vbase_p + (size_t)(kb0 + 128) * HEAD_DIM,
                       HEAD_DIM, 128, HEAD_DIM, SEQ, HEAD_DIM);
    }
#endif
    { // stage this block's mask tile (128x128 bytes) with b128 loads
      const int row  = tid >> 1;
      const int coff = (tid & 1) * 64;
      const unsigned char* srcb =
          mask + ((size_t)b * SEQ + q0 + row) * SEQ + kb0 + coff;
      const uint4* src = (const uint4*)srcb;
      uint4* dst = (uint4*)&Ms[row][coff];
#pragma unroll
      for (int j = 0; j < 4; ++j) dst[j] = src[j];
      if (i + 1 < NKB) __builtin_prefetch(srcb + 128, 0, 0);  // next mask tile
    }
    __syncthreads();   // mask visible; K/V[cur] buffer already published

    // ---- scores: S = Q(16x64) @ K^T(64x128), 16 WMMAs ----
    v8f s[8];
#pragma unroll
    for (int nt = 0; nt < 8; ++nt) s[nt] = vzero;
#pragma unroll
    for (int d0i = 0; d0i < 2; ++d0i) {
      const int d0 = d0i * 32;
#pragma unroll
      for (int nt = 0; nt < 8; ++nt) {
        v16bf bk;
        const int key = nt * 16 + l15;
#pragma unroll
        for (int e = 0; e < 16; ++e) bk[e] = Ks[cur][key][d0 + kbb + e];
        s[nt] = __builtin_amdgcn_wmma_f32_16x16x32_bf16(
            false, aq[d0i], false, bk, (short)0, s[nt], false, false);
      }
    }

    // ---- scale 1/sqrt(64) + boolean mask (from LDS) ----
#pragma unroll
    for (int nt = 0; nt < 8; ++nt) {
#pragma unroll
      for (int r = 0; r < 8; ++r) {
        const unsigned char mm = Ms[wave * 16 + r + half8][nt * 16 + l15];
        s[nt][r] = mm ? -1e30f : s[nt][r] * 0.125f;
      }
    }

    // ---- online softmax stats (rows live in 16-lane halves) ----
    float mnew[8], alpha[8], psum[8];
#pragma unroll
    for (int r = 0; r < 8; ++r) {
      float rm = s[0][r];
#pragma unroll
      for (int nt = 1; nt < 8; ++nt) rm = fmaxf(rm, s[nt][r]);
      rm = half16_max(rm);
      mnew[r]  = fmaxf(m_run[r], rm);
      alpha[r] = __expf(m_run[r] - mnew[r]);
      m_run[r] = mnew[r];
      psum[r]  = 0.f;
    }
#pragma unroll
    for (int nt = 0; nt < 8; ++nt) {
#pragma unroll
      for (int r = 0; r < 8; ++r) {
        const float p = __expf(s[nt][r] - mnew[r]);
        psum[r] += p;
        Ps[wave][r + half8][nt * 16 + l15] = f2bf(p);
      }
    }
#pragma unroll
    for (int r = 0; r < 8; ++r) {
      l_run[r] = l_run[r] * alpha[r] + half16_sum(psum[r]);
#pragma unroll
      for (int nt = 0; nt < 4; ++nt) ctx[nt][r] *= alpha[r];
    }

    // ---- context += P(16x128) @ V(128x64), 16 WMMAs (Ps is wave-private) ----
#pragma unroll
    for (int kc = 0; kc < 4; ++kc) {
      v16bf ap;
#pragma unroll
      for (int e = 0; e < 8; ++e) {
        ap[e]     = Ps[wave][l15][kc * 32 + kbb + e];
        ap[8 + e] = Ps[wave][l15][kc * 32 + kbb + 8 + e];
      }
#pragma unroll
      for (int nt = 0; nt < 4; ++nt) {
        v16bf bv;
        const int d = nt * 16 + l15;
#pragma unroll
        for (int e = 0; e < 16; ++e) bv[e] = Vs[cur][kc * 32 + kbb + e][d];
        ctx[nt] = __builtin_amdgcn_wmma_f32_16x16x32_bf16(
            false, ap, false, bv, (short)0, ctx[nt], false, false);
      }
    }

    if (i + 1 < NKB) {
#if HAVE_TDM
      if (wave == 0) __builtin_amdgcn_s_wait_tensorcnt(0);
#else
      { // cooperative staging of next K/V block into the other buffer
        const int row  = tid >> 1;
        const int doff = (tid & 1) * 32;
        const size_t base = (size_t)(kb0 + 128 + row) * HEAD_DIM + doff;
        stage16(&Ks[1 - cur][row][doff],      kbase_p + base);
        stage16(&Ks[1 - cur][row][doff + 16], kbase_p + base + 16);
        stage16(&Vs[1 - cur][row][doff],      vbase_p + base);
        stage16(&Vs[1 - cur][row][doff + 16], vbase_p + base + 16);
      }
#endif
      __syncthreads();   // publish next buffer; fence Ms reads vs next writes
    }
  }

  // ---- normalize and write context as [b, s, h*64+d] bf16 ----
#pragma unroll
  for (int nt = 0; nt < 4; ++nt) {
    const int d = nt * 16 + l15;
#pragma unroll
    for (int r = 0; r < 8; ++r) {
      const int row = q0 + wave * 16 + r + half8;
      const float inv = 1.0f / l_run[r];
      ctxw[((size_t)b * SEQ + row) * D_MODEL + h * HEAD_DIM + d] =
          f2bf(ctx[nt][r] * inv);
    }
  }
}

// ---------------------------------------------------------------------------
extern "C" void kernel_launch(void* const* d_in, const int* in_sizes, int n_in,
                              void* d_out, int out_size, void* d_ws, size_t ws_size,
                              hipStream_t stream) {
  const float* Q  = (const float*)d_in[0];
  const float* K  = (const float*)d_in[1];
  const float* V  = (const float*)d_in[2];
  const unsigned char* mask = (const unsigned char*)d_in[3];  // numpy bool = 1B
  const float* Wq = (const float*)d_in[4];
  const float* bq = (const float*)d_in[5];
  const float* Wk = (const float*)d_in[6];
  const float* bk = (const float*)d_in[7];
  const float* Wv = (const float*)d_in[8];
  const float* bv = (const float*)d_in[9];
  const float* Wo = (const float*)d_in[10];
  const float* bo = (const float*)d_in[11];

  const size_t HELEMS = (size_t)BATCH * N_HEADS * SEQ * HEAD_DIM; // 4,194,304
  bf16* qb = (bf16*)d_ws;
  bf16* kb = qb + HELEMS;
  bf16* vb = kb + HELEMS;
  bf16* cb = vb + HELEMS;   // [B, S, D_MODEL] bf16 context; total ws: 32 MiB

  const int M = BATCH * SEQ;              // 4096
  dim3 blk(256);
  dim3 gGemm(D_MODEL / 128, M / 128);     // (8, 32)
  dim3 gAttn(SEQ / 128, BATCH * N_HEADS); // (16, 32)

  gemm_bias_wmma<float, 0><<<gGemm, blk, 0, stream>>>(Q, Wq, bq, qb, nullptr, M, D_MODEL, D_MODEL);
  gemm_bias_wmma<float, 0><<<gGemm, blk, 0, stream>>>(K, Wk, bk, kb, nullptr, M, D_MODEL, D_MODEL);
  gemm_bias_wmma<float, 0><<<gGemm, blk, 0, stream>>>(V, Wv, bv, vb, nullptr, M, D_MODEL, D_MODEL);
  attention_wmma<<<gAttn, blk, 0, stream>>>(qb, kb, vb, mask, cb);
  gemm_bias_wmma<bf16, 1><<<gGemm, blk, 0, stream>>>(cb, Wo, bo, nullptr, (float*)d_out, M, D_MODEL, D_MODEL);
}
